// MHA_19035295056060
// MI455X (gfx1250) — matrix operand order repaired and checked by reading, not compiled, not run on hardware
//
#include <hip/hip_runtime.h>
#include <hip/hip_bf16.h>
#include <stdint.h>

typedef __bf16 bf16_t;
typedef __attribute__((ext_vector_type(16))) __bf16 bf16x16;
typedef __attribute__((ext_vector_type(8)))  __bf16 bf16x8;
typedef __attribute__((ext_vector_type(8)))  float  f32x8;

#define S_LEN   2048
#define D_MODEL 2048
#define N_HEADS 16
#define HEAD_DIM 128
#define BATCH   2

union FragU { bf16x16 v; bf16x8 h[2]; };

__device__ __forceinline__ f32x8 wmma_bf16(bf16x16 a, bf16x16 b, f32x8 c) {
    return __builtin_amdgcn_wmma_f32_16x16x32_bf16(false, a, false, b, (short)0, c, false, false);
}

// ---- CDNA5 async memory->LDS copy (ASYNCcnt-tracked, no VGPR data path) ----
// GLOBAL_LOAD_ASYNC_TO_LDS_B128, GV mode: vdst = LDS byte address, vaddr = 64b global address.
__device__ __forceinline__ void async_ld16(uint32_t lds_off, const bf16_t* g) {
    asm volatile("global_load_async_to_lds_b128 %0, %1, off"
                 :: "v"(lds_off), "v"(g) : "memory");
}
__device__ __forceinline__ void wait_async0() {
    asm volatile("s_wait_asynccnt 0x0" ::: "memory");
}
__device__ __forceinline__ uint32_t lds_addr(const void* p) {
    return (uint32_t)(uintptr_t)p;   // LDS aperture: low 32 bits = LDS byte offset
}

// A fragment: 16 rows x 32 K from row-major [row][K].
// lanes 0-15: K 0..7 / 16..23 ; lanes 16-31: K 8..15 / 24..31 (ISA 16-bit A layout).
__device__ __forceinline__ bf16x16 load_a_frag(const bf16_t* base, int stride, int m0, int k0, int lane) {
    int half = lane >> 4, r = lane & 15;
    const bf16_t* p = base + (size_t)(m0 + r) * stride;
    FragU f;
    f.h[0] = *(const bf16x8*)(p + k0 + half * 8);
    f.h[1] = *(const bf16x8*)(p + k0 + 16 + half * 8);
    return f.v;
}

// B fragment: 32 K x 16 cols from column-major source stored as Bt[N][K].
// lanes 0-15: col N, K 0..7 / 16..23 ; lanes 16-31: col N, K 8..15 / 24..31.
__device__ __forceinline__ bf16x16 load_b_frag(const bf16_t* base, int stride, int n0, int k0, int lane) {
    int half = lane >> 4, c = lane & 15;
    const bf16_t* p = base + (size_t)(n0 + c) * stride + k0 + half * 8;
    FragU f;
    f.h[0] = *(const bf16x8*)(p);
    f.h[1] = *(const bf16x8*)(p + 16);
    return f.v;
}

// ---------------- conversion kernels ----------------
__global__ void cvt_f32_bf16(const float* __restrict__ in, bf16_t* __restrict__ out, int n) {
    int i = blockIdx.x * 256 + threadIdx.x;
    if (i < n) out[i] = (bf16_t)in[i];
}

__global__ void transpose_f32_bf16(const float* __restrict__ in, bf16_t* __restrict__ out,
                                   int rows, int cols) {
    int i = blockIdx.x * 256 + threadIdx.x;
    if (i < rows * cols) {
        int r = i / cols, c = i - r * cols;
        out[(size_t)c * rows + r] = (bf16_t)in[i];
    }
}

// ---------------- GEMM: C[M,N] = A[M,K](bf16) * Bt[N,K]^T + bias ----------------
// 256 threads (8 waves), block tile 128x128, wave tile 32x64, K-step 32.
// Double-buffered LDS fed by async global->LDS DMA, pipelined one K-step ahead.
template <typename OutT>
__global__ void __launch_bounds__(256)
gemm_wmma(const bf16_t* __restrict__ A, const bf16_t* __restrict__ Bt,
          const float* __restrict__ bias, OutT* __restrict__ C,
          int M, int N, int K) {
    __shared__ bf16_t As[2][128][40];   // 80B row stride (16B multiple)
    __shared__ bf16_t Bs[2][128][40];
    const int tid  = threadIdx.x;
    const int wave = tid >> 5, lane = tid & 31;
    const int half = lane >> 4, ln = lane & 15;
    const int wr = wave >> 1, wc = wave & 1;       // 4x2 wave grid
    const int mBase = blockIdx.y * 128;
    const int nBase = blockIdx.x * 128;

    const int sRow  = tid >> 1;          // 0..127
    const int sKoff = (tid & 1) * 16;    // 0 or 16
    const bf16_t* gA = A  + (size_t)(mBase + sRow) * K + sKoff;
    const bf16_t* gB = Bt + (size_t)(nBase + sRow) * K + sKoff;

    auto stage = [&](int buf, int k0) {
        uint32_t la = lds_addr(&As[buf][sRow][sKoff]);
        uint32_t lb = lds_addr(&Bs[buf][sRow][sKoff]);
        async_ld16(la,      gA + k0);
        async_ld16(la + 16, gA + k0 + 8);
        async_ld16(lb,      gB + k0);
        async_ld16(lb + 16, gB + k0 + 8);
    };

    f32x8 acc[2][4];
#pragma unroll
    for (int f = 0; f < 2; ++f)
#pragma unroll
        for (int c = 0; c < 4; ++c)
#pragma unroll
            for (int r = 0; r < 8; ++r) acc[f][c][r] = 0.0f;

    const int iters = K / 32;
    stage(0, 0);
    for (int it = 0; it < iters; ++it) {
        int cur = it & 1;
        wait_async0();          // my stage of buf `cur` (issued last iter) done
        __syncthreads();        // everyone's stage done; all done reading buf cur^1
        if (it + 1 < iters) stage(cur ^ 1, (it + 1) * 32);   // DMA overlaps compute
        bf16x16 aF0 = load_a_frag(&As[cur][0][0], 40, wr * 32,      0, lane);
        bf16x16 aF1 = load_a_frag(&As[cur][0][0], 40, wr * 32 + 16, 0, lane);
#pragma unroll
        for (int c = 0; c < 4; ++c) {
            bf16x16 bF = load_b_frag(&Bs[cur][0][0], 40, wc * 64 + c * 16, 0, lane);
            acc[0][c] = wmma_bf16(aF0, bF, acc[0][c]);
            acc[1][c] = wmma_bf16(aF1, bF, acc[1][c]);
        }
    }

    // epilogue: C/D layout -> (M = r + half*8, N = ln)
#pragma unroll
    for (int c = 0; c < 4; ++c) {
        int n = nBase + wc * 64 + c * 16 + ln;
        float bv = bias[n];
#pragma unroll
        for (int f = 0; f < 2; ++f)
#pragma unroll
            for (int r = 0; r < 8; ++r) {
                int m = mBase + wr * 32 + f * 16 + r + half * 8;
                C[(size_t)m * N + n] = (OutT)(acc[f][c][r] + bv);
            }
    }
}

// ---------------- RoPE + head split ----------------
__global__ void __launch_bounds__(128)
rope_split(const bf16_t* __restrict__ qkv, bf16_t* __restrict__ Q,
           bf16_t* __restrict__ Kd, bf16_t* __restrict__ Vt) {
    int idx = blockIdx.x;
    int s = idx & (S_LEN - 1);
    int h = (idx >> 11) & (N_HEADS - 1);
    int b = idx >> 15;
    int t = threadIdx.x;
    const bf16_t* src = qkv + (size_t)(b * S_LEN + s) * (3 * D_MODEL);
    int bh = b * N_HEADS + h;

    Vt[((size_t)bh * HEAD_DIM + t) * S_LEN + s] = src[2 * D_MODEL + h * HEAD_DIM + t];

    if (t < 64) {
        float theta = __powf(10000.0f, -((float)(2 * t) / (float)HEAD_DIM));
        float ang = (float)s * theta;
        float cs = __cosf(ang), sn = __sinf(ang);
        const float sc = 0.08838834764831845f;   // 1/sqrt(128), folded into Q

        float x1 = (float)src[h * HEAD_DIM + 2 * t];
        float x2 = (float)src[h * HEAD_DIM + 2 * t + 1];
        bf16_t* qd = Q + ((size_t)bh * S_LEN + s) * HEAD_DIM;
        qd[t]      = (bf16_t)((x1 * cs - x2 * sn) * sc);
        qd[t + 64] = (bf16_t)((x1 * sn + x2 * cs) * sc);

        float y1 = (float)src[D_MODEL + h * HEAD_DIM + 2 * t];
        float y2 = (float)src[D_MODEL + h * HEAD_DIM + 2 * t + 1];
        bf16_t* kd = Kd + ((size_t)bh * S_LEN + s) * HEAD_DIM;
        kd[t]      = (bf16_t)(y1 * cs - y2 * sn);
        kd[t + 64] = (bf16_t)(y1 * sn + y2 * cs);
    }
}

// ---------------- flash attention ----------------
// grid (B*H, S/64); 128 threads = 4 waves; wave owns 16 query rows.
// Double-buffered async K/V tile staging pipelined one 32-key step ahead.
__global__ void __launch_bounds__(128)
attention(const bf16_t* __restrict__ Q, const bf16_t* __restrict__ Kd,
          const bf16_t* __restrict__ Vt, bf16_t* __restrict__ Ao) {
    __shared__ bf16_t Ks[2][32][136];     // keys x dims, 272B row stride
    __shared__ bf16_t Vs[2][128][40];     // dims x keys, 80B row stride
    __shared__ bf16_t Ps[4][16][40];      // per-wave P staging (16x32)

    const int bh   = blockIdx.x;
    const int tid  = threadIdx.x;
    const int wave = tid >> 5, lane = tid & 31;
    const int half = lane >> 4, ln = lane & 15;
    const int q0   = blockIdx.y * 64 + wave * 16;

    const bf16_t* Qbh = Q  + (size_t)bh * S_LEN * HEAD_DIM;
    const bf16_t* Kbh = Kd + (size_t)bh * S_LEN * HEAD_DIM;
    const bf16_t* Vbh = Vt + (size_t)bh * HEAD_DIM * S_LEN;

    const int kRow = tid >> 2, kOff = (tid & 3) * 32;
    auto stageKV = [&](int buf, int kt) {
        const bf16_t* gk = Kbh + (size_t)(kt + kRow) * HEAD_DIM + kOff;
        uint32_t lk = lds_addr(&Ks[buf][kRow][kOff]);
        async_ld16(lk,      gk);      async_ld16(lk + 16, gk + 8);
        async_ld16(lk + 32, gk + 16); async_ld16(lk + 48, gk + 24);
        const bf16_t* gv = Vbh + (size_t)tid * S_LEN + kt;
        uint32_t lv = lds_addr(&Vs[buf][tid][0]);
        async_ld16(lv,      gv);      async_ld16(lv + 16, gv + 8);
        async_ld16(lv + 32, gv + 16); async_ld16(lv + 48, gv + 24);
    };

    bf16x16 qf[4];
#pragma unroll
    for (int i = 0; i < 4; ++i) qf[i] = load_a_frag(Qbh, HEAD_DIM, q0, i * 32, lane);

    f32x8 o[8];
    float m[8], l[8];
#pragma unroll
    for (int r = 0; r < 8; ++r) { m[r] = -1e30f; l[r] = 0.0f; }
#pragma unroll
    for (int d = 0; d < 8; ++d)
#pragma unroll
        for (int r = 0; r < 8; ++r) o[d][r] = 0.0f;

    const int iters = S_LEN / 32;
    stageKV(0, 0);
    for (int it = 0; it < iters; ++it) {
        int cur = it & 1;
        wait_async0();
        __syncthreads();
        if (it + 1 < iters) stageKV(cur ^ 1, (it + 1) * 32);

        // scores: two 16x16 tiles over the 128-dim contraction
        f32x8 s0, s1;
#pragma unroll
        for (int r = 0; r < 8; ++r) { s0[r] = 0.0f; s1[r] = 0.0f; }
#pragma unroll
        for (int i = 0; i < 4; ++i) {
            bf16x16 b0 = load_b_frag(&Ks[cur][0][0], 136, 0,  i * 32, lane);
            bf16x16 b1 = load_b_frag(&Ks[cur][0][0], 136, 16, i * 32, lane);
            s0 = wmma_bf16(qf[i], b0, s0);
            s1 = wmma_bf16(qf[i], b1, s1);
        }

        // online softmax; rows live across the 16-lane half
        float alpha[8];
#pragma unroll
        for (int r = 0; r < 8; ++r) {
            float rm = fmaxf(s0[r], s1[r]);
#pragma unroll
            for (int off = 8; off >= 1; off >>= 1) rm = fmaxf(rm, __shfl_xor(rm, off, 16));
            float mn = fmaxf(m[r], rm);
            alpha[r] = __expf(m[r] - mn);
            float p0 = __expf(s0[r] - mn);
            float p1 = __expf(s1[r] - mn);
            float rs = p0 + p1;
#pragma unroll
            for (int off = 8; off >= 1; off >>= 1) rs += __shfl_xor(rs, off, 16);
            l[r] = l[r] * alpha[r] + rs;
            m[r] = mn;
            s0[r] = p0;
            s1[r] = p1;
        }
#pragma unroll
        for (int d = 0; d < 8; ++d)
#pragma unroll
            for (int r = 0; r < 8; ++r) o[d][r] *= alpha[r];

        // P: C-layout -> A-fragment via private LDS patch (same-wave DS is in-order)
#pragma unroll
        for (int r = 0; r < 8; ++r) {
            Ps[wave][r + half * 8][ln]      = (bf16_t)s0[r];
            Ps[wave][r + half * 8][16 + ln] = (bf16_t)s1[r];
        }
        bf16x16 pf = load_a_frag(&Ps[wave][0][0], 40, 0, 0, lane);
#pragma unroll
        for (int d = 0; d < 8; ++d) {
            bf16x16 vf = load_b_frag(&Vs[cur][0][0], 40, d * 16, 0, lane);
            o[d] = wmma_bf16(pf, vf, o[d]);
        }
    }

    // epilogue: Ao[b*S + q][h*128 + dim]
    int b = bh >> 4, h = bh & 15;
#pragma unroll
    for (int r = 0; r < 8; ++r) {
        float inv = 1.0f / l[r];
        size_t rowoff = ((size_t)(b * S_LEN + q0 + r + half * 8)) * D_MODEL + h * HEAD_DIM;
#pragma unroll
        for (int d = 0; d < 8; ++d)
            Ao[rowoff + d * 16 + ln] = (bf16_t)(o[d][r] * inv);
    }
}

// ---------------- host launch ----------------
extern "C" void kernel_launch(void* const* d_in, const int* in_sizes, int n_in,
                              void* d_out, int out_size, void* d_ws, size_t ws_size,
                              hipStream_t stream) {
    const float* x    = (const float*)d_in[0];
    const float* Wqkv = (const float*)d_in[1];
    const float* bqkv = (const float*)d_in[2];
    const float* Wo   = (const float*)d_in[3];
    const float* bo   = (const float*)d_in[4];
    float* out = (float*)d_out;

    char* ws = (char*)d_ws;
    const size_t BS = (size_t)BATCH * S_LEN;          // 4096
    bf16_t* xb   = (bf16_t*)(ws);                     // [4096,2048]
    bf16_t* Wt   = (bf16_t*)(ws +  16777216);         // [6144,2048]
    bf16_t* Wot  = (bf16_t*)(ws +  41943040);         // [2048,2048]
    bf16_t* qkv  = (bf16_t*)(ws +  50331648);         // [4096,6144]
    bf16_t* Qb   = (bf16_t*)(ws + 100663296);         // [32,2048,128]
    bf16_t* Kb   = (bf16_t*)(ws + 117440512);         // [32,2048,128]
    bf16_t* Vtb  = (bf16_t*)(ws + 134217728);         // [32,128,2048]
    bf16_t* Ao   = (bf16_t*)(ws + 150994944);         // [4096,2048]

    cvt_f32_bf16<<<(int)((BS * D_MODEL) / 256), 256, 0, stream>>>(x, xb, (int)(BS * D_MODEL));
    transpose_f32_bf16<<<(D_MODEL * 3 * D_MODEL) / 256, 256, 0, stream>>>(Wqkv, Wt, D_MODEL, 3 * D_MODEL);
    transpose_f32_bf16<<<(D_MODEL * D_MODEL) / 256, 256, 0, stream>>>(Wo, Wot, D_MODEL, D_MODEL);

    gemm_wmma<bf16_t><<<dim3(3 * D_MODEL / 128, (int)(BS / 128)), 256, 0, stream>>>(
        xb, Wt, bqkv, qkv, (int)BS, 3 * D_MODEL, D_MODEL);

    rope_split<<<BATCH * N_HEADS * S_LEN, 128, 0, stream>>>(qkv, Qb, Kb, Vtb);

    attention<<<dim3(BATCH * N_HEADS, S_LEN / 64), 128, 0, stream>>>(Qb, Kb, Vtb, Ao);

    gemm_wmma<float><<<dim3(D_MODEL / 128, (int)(BS / 128)), 256, 0, stream>>>(
        Ao, Wot, bo, out, (int)BS, D_MODEL, D_MODEL);
}
